// GENnopos_14087492730942
// MI455X (gfx1250) — compile-verified, matrix operand order checked by reading, not run on hardware
//
#include <hip/hip_runtime.h>

// ---------------------------------------------------------------------------
// CDNA5 (gfx1250) wave32 WMMA implementation of the graph-neural-process
// forward pass.  All K>=32 GEMMs run on v_wmma_f32_16x16x32_f16 (f16 in,
// f32 accumulate).  B operand staged in LDS in WMMA-swizzled f16 layout
// (one ds_load_b128 pair per 32x16 tile per lane).  Epilogues fused.
// ---------------------------------------------------------------------------

typedef __attribute__((ext_vector_type(16))) _Float16 v16h;
typedef __attribute__((ext_vector_type(8)))  _Float16 v8h;
typedef __attribute__((ext_vector_type(4)))  _Float16 v4h;
typedef __attribute__((ext_vector_type(8)))  float    v8f;
typedef __attribute__((ext_vector_type(4)))  float    f32x4;

#define H_DIM 128
#define EPSLN 1e-5f
#define KSLAB 128            // B-slab depth staged in LDS (32 KB)

// ---------------- A-operand functors (global, load 8 k-consecutive) ---------

struct AMatF32 {                 // row-major fp32 [M][ld]
    const float* p; int ld;
    __device__ void load8(int m, int k0, _Float16* d) const {
        const f32x4* q = (const f32x4*)(p + (size_t)m * ld + k0);
        f32x4 a = q[0], b = q[1];
        d[0]=(_Float16)a.x; d[1]=(_Float16)a.y; d[2]=(_Float16)a.z; d[3]=(_Float16)a.w;
        d[4]=(_Float16)b.x; d[5]=(_Float16)b.y; d[6]=(_Float16)b.z; d[7]=(_Float16)b.w;
    }
};
struct AConcat2F32 {             // concat(p0, p1) along K, each [M][ld] (ld=128)
    const float* p0; const float* p1; int ld;
    __device__ void load8(int m, int k0, _Float16* d) const {
        const float* src = (k0 < ld) ? (p0 + (size_t)m * ld + k0)
                                     : (p1 + (size_t)m * ld + (k0 - ld));
        const f32x4* q = (const f32x4*)src;
        f32x4 a = q[0], b = q[1];
        d[0]=(_Float16)a.x; d[1]=(_Float16)a.y; d[2]=(_Float16)a.z; d[3]=(_Float16)a.w;
        d[4]=(_Float16)b.x; d[5]=(_Float16)b.y; d[6]=(_Float16)b.z; d[7]=(_Float16)b.w;
    }
};
struct AEdgeConcat {             // row e = concat(nodes[recv[e]], nodes[send[e]])
    const float* nodes; const int* recv; const int* send;
    __device__ void load8(int e, int k0, _Float16* d) const {
        int node = (k0 < H_DIM) ? recv[e] : send[e];
        int kk   = (k0 < H_DIM) ? k0 : k0 - H_DIM;
        const f32x4* q = (const f32x4*)(nodes + (size_t)node * H_DIM + kk);
        f32x4 a = q[0], b = q[1];
        d[0]=(_Float16)a.x; d[1]=(_Float16)a.y; d[2]=(_Float16)a.z; d[3]=(_Float16)a.w;
        d[4]=(_Float16)b.x; d[5]=(_Float16)b.y; d[6]=(_Float16)b.z; d[7]=(_Float16)b.w;
    }
};
struct AScT {                    // a(m=n, k=c) = sc[c][n]  (true transpose gather)
    const _Float16* sc; int N;
    __device__ void load8(int n, int c0, _Float16* d) const {
#pragma unroll
        for (int i = 0; i < 8; ++i) d[i] = sc[(size_t)(c0 + i) * N + n];
    }
};
struct AF16Row {                 // row-major f16 [M][ld]
    const _Float16* p; int ld;
    __device__ void load8(int m, int k0, _Float16* d) const {
        v8h v = *(const v8h*)(p + (size_t)m * ld + k0);
#pragma unroll
        for (int i = 0; i < 8; ++i) d[i] = v[i];
    }
};

// ---------------- B-operand functors (load 4 n-consecutive, fixed k) --------

struct BF32Row {                 // row-major fp32 [K][128]
    const float* w;
    __device__ f32x4 load4(int k, int n0) const {
        return *(const f32x4*)(w + (size_t)k * H_DIM + n0);
    }
};
struct BF16Row {                 // row-major f16 [K][128]
    const _Float16* w;
    __device__ f32x4 load4(int k, int n0) const {
        v4h v = *(const v4h*)(w + (size_t)k * H_DIM + n0);
        f32x4 r; r.x=(float)v[0]; r.y=(float)v[1]; r.z=(float)v[2]; r.w=(float)v[3];
        return r;
    }
};

// ---------------- epilogue description --------------------------------------

struct Epi {
    const float* bias;      // nullable
    const float* ln_g;      // non-null => layernorm over 128 cols
    const float* ln_b;
    float*       out_f32;   // [M][ldo]  (nullable)
    _Float16*    out_f16;   // [M][ldo]  (nullable)
    int          ldo;
    int          relu;
    const int*   scat_idx;  // non-null => atomicAdd into scat_out[idx[row]*128+col]
    float*       scat_out;
};

// ---------------- generic WMMA GEMM (N fixed = 128) --------------------------
// one wave computes a 16x128 tile of C; block = 8 waves = 128 rows.
// B staged per KSLAB in LDS, swizzled [chunk][tile][lane][16 halves] so the
// wave loop reads each 32x16 B tile with one contiguous 32B LDS read per lane.
// Requires M % 128 == 0, K % 128 == 0 (true for every call site).

template <class AFn, class BFn>
__global__ __launch_bounds__(256) void gemm_wmma(AFn A, BFn Bf, int M, int K, Epi e) {
    __shared__ _Float16 lB[KSLAB * H_DIM];          // 32 KB

    const int wave = threadIdx.x >> 5;
    const int lane = threadIdx.x & 31;
    const int g    = lane >> 4;          // lane group (0/1)
    const int r    = lane & 15;          // row (A) / col (B,C) within group
    const int row0 = (blockIdx.x * 8 + wave) * 16;  // always < M (M % 128 == 0)

    v8f acc[8];
#pragma unroll
    for (int t = 0; t < 8; ++t)
#pragma unroll
        for (int v = 0; v < 8; ++v) acc[t][v] = 0.f;

    for (int ks = 0; ks < K; ks += KSLAB) {
        const int kslab = (K - ks < KSLAB) ? (K - ks) : KSLAB;

        __syncthreads();                 // previous slab fully consumed
        // ---- cooperative staging of B slab into swizzled LDS ----
        for (int task = threadIdx.x; task < kslab * 32; task += 256) {
            int k  = task >> 5;                      // 0..kslab-1
            int n0 = (task & 31) << 2;               // 0,4,...,124
            f32x4 w4 = Bf.load4(ks + k, n0);
            int chunk = k >> 5;
            int kk    = k & 31;
            int g2    = (kk >> 3) & 1;
            int eoff  = ((kk >> 4) << 3) | (kk & 7);
            float vals[4] = {w4.x, w4.y, w4.z, w4.w};
#pragma unroll
            for (int j = 0; j < 4; ++j) {
                int n   = n0 + j;
                int idx = ((chunk * 8 + (n >> 4)) * 32 + g2 * 16 + (n & 15)) * 16 + eoff;
                lB[idx] = (_Float16)vals[j];
            }
        }
        __syncthreads();

        // ---- WMMA over this slab ----
        for (int kc = 0; kc < kslab; kc += 32) {
            v16h a;
            _Float16 tmp[8];
            A.load8(row0 + r, ks + kc + 8 * g, tmp);
#pragma unroll
            for (int i = 0; i < 8; ++i) a[i] = tmp[i];
            A.load8(row0 + r, ks + kc + 16 + 8 * g, tmp);
#pragma unroll
            for (int i = 0; i < 8; ++i) a[8 + i] = tmp[i];

            const int cbase = (kc >> 5) * 8;
#pragma unroll
            for (int t = 0; t < 8; ++t) {
                v16h bb = *(const v16h*)&lB[(((cbase + t) * 32) + lane) << 4];
                acc[t] = __builtin_amdgcn_wmma_f32_16x16x32_f16(
                    false, a, false, bb, (short)0, acc[t], false, false);
            }
        }
    }

    // ---- bias + relu ----
#pragma unroll
    for (int t = 0; t < 8; ++t) {
        int col  = t * 16 + r;
        float bi = e.bias ? e.bias[col] : 0.f;
#pragma unroll
        for (int v = 0; v < 8; ++v) {
            float x = acc[t][v] + bi;
            if (e.relu) x = fmaxf(x, 0.f);
            acc[t][v] = x;
        }
    }

    // ---- layernorm over the 128 columns (row m = v + 8g in a 16-lane group) ----
    if (e.ln_g) {
#pragma unroll
        for (int v = 0; v < 8; ++v) {
            float s = 0.f, s2 = 0.f;
#pragma unroll
            for (int t = 0; t < 8; ++t) { float x = acc[t][v]; s += x; s2 += x * x; }
            for (int off = 1; off < 16; off <<= 1) {
                s  += __shfl_xor(s,  off, 32);
                s2 += __shfl_xor(s2, off, 32);
            }
            float mu  = s * (1.f / 128.f);
            float var = s2 * (1.f / 128.f) - mu * mu;
            float inv = rsqrtf(var + EPSLN);
#pragma unroll
            for (int t = 0; t < 8; ++t) {
                int col   = t * 16 + r;
                acc[t][v] = (acc[t][v] - mu) * inv * e.ln_g[col] + e.ln_b[col];
            }
        }
    }

    // ---- store / scatter ----
#pragma unroll
    for (int t = 0; t < 8; ++t) {
        int col = t * 16 + r;
#pragma unroll
        for (int v = 0; v < 8; ++v) {
            int   row = row0 + v + 8 * g;
            float x   = acc[t][v];
            if (e.scat_idx) {
                atomicAdd(&e.scat_out[(size_t)e.scat_idx[row] * H_DIM + col], x);
            } else if (e.out_f16) {
                e.out_f16[(size_t)row * e.ldo + col] = (_Float16)x;
            } else {
                e.out_f32[(size_t)row * e.ldo + col] = x;
            }
        }
    }
}

// ---------------- softmax(-||x - pos||^2) over 1024 nodes --------------------

__global__ __launch_bounds__(256) void scores_kernel(const float* __restrict__ x,
                                                     const float* __restrict__ pos,
                                                     _Float16* __restrict__ out) {
    __shared__ float2 spos[1024];
    for (int i = threadIdx.x; i < 1024; i += 256)
        spos[i] = ((const float2*)pos)[i];
    __syncthreads();

    const int wave = threadIdx.x >> 5;
    const int lane = threadIdx.x & 31;
    const size_t pt = (size_t)blockIdx.x * 8 + wave;
    float2 xp = ((const float2*)x)[pt];

    float d2v[32];
    float mn = 3.4e38f;
#pragma unroll
    for (int i = 0; i < 32; ++i) {
        int n    = i * 32 + lane;
        float dx = xp.x - spos[n].x, dy = xp.y - spos[n].y;
        float d2 = dx * dx + dy * dy;
        d2v[i] = d2;
        mn = fminf(mn, d2);
    }
    for (int off = 1; off < 32; off <<= 1) mn = fminf(mn, __shfl_xor(mn, off, 32));
    float sum = 0.f;
#pragma unroll
    for (int i = 0; i < 32; ++i) { float ex = __expf(mn - d2v[i]); d2v[i] = ex; sum += ex; }
    for (int off = 1; off < 32; off <<= 1) sum += __shfl_xor(sum, off, 32);
    float inv = 1.f / sum;
#pragma unroll
    for (int i = 0; i < 32; ++i)
        out[pt * 1024 + i * 32 + lane] = (_Float16)(d2v[i] * inv);
}

// ---------------- tiny-K first MLP layer (K = da+db <= 5), relu --------------

__global__ __launch_bounds__(256) void mlp0_kernel(const float* __restrict__ xa, int da,
                                                   const float* __restrict__ xb, int db,
                                                   const float* __restrict__ W,
                                                   const float* __restrict__ bias,
                                                   float* __restrict__ out, int M) {
    int idx = blockIdx.x * 256 + threadIdx.x;
    int row = idx >> 7, h = idx & 127;
    if (row >= M) return;
    float acc = bias[h];
    for (int k = 0; k < da; ++k) acc += xa[(size_t)row * da + k] * W[k * H_DIM + h];
    for (int k = 0; k < db; ++k) acc += xb[(size_t)row * db + k] * W[(da + k) * H_DIM + h];
    out[(size_t)row * H_DIM + h] = fmaxf(acc, 0.f);
}

// ---------------- decoder tail: [M][128] @ [128][3] + b ----------------------

__global__ __launch_bounds__(256) void dec_final_kernel(const float* __restrict__ hin,
                                                        const float* __restrict__ W,
                                                        const float* __restrict__ bias,
                                                        float* __restrict__ out, int M) {
    int row = blockIdx.x * 256 + threadIdx.x;
    if (row >= M) return;
    float a0 = bias[0], a1 = bias[1], a2 = bias[2];
    for (int k = 0; k < 128; ++k) {
        float x = hin[(size_t)row * 128 + k];
        a0 += x * W[k * 3 + 0];
        a1 += x * W[k * 3 + 1];
        a2 += x * W[k * 3 + 2];
    }
    out[(size_t)row * 3 + 0] = a0;
    out[(size_t)row * 3 + 1] = a1;
    out[(size_t)row * 3 + 2] = a2;
}

__global__ __launch_bounds__(256) void zero_f32_kernel(float* p, int n) {
    int i = blockIdx.x * 256 + threadIdx.x;
    if (i < n) p[i] = 0.f;
}

// ---------------- host side --------------------------------------------------

template <class AFn, class BFn>
static void launch_gemm(hipStream_t s, AFn A, BFn Bf, int M, int K, Epi e) {
    gemm_wmma<AFn, BFn><<<dim3(M / 128), dim3(256), 0, s>>>(A, Bf, M, K, e);
}

static Epi epi_plain(float* o, int ldo) { return Epi{nullptr, nullptr, nullptr, o, nullptr, ldo, 0, nullptr, nullptr}; }

extern "C" void kernel_launch(void* const* d_in, const int* in_sizes, int n_in,
                              void* d_out, int out_size, void* d_ws, size_t ws_size,
                              hipStream_t stream) {
    (void)in_sizes; (void)n_in; (void)out_size; (void)ws_size;

    const size_t B  = 8, Nc = 4096, Nt = 4096, Nn = 1024, E = 16384;
    const size_t MR = B * Nc;            // 32768 rows (context == target count)

    const float* xc      = (const float*)d_in[0];
    const float* yc      = (const float*)d_in[1];
    const float* xt      = (const float*)d_in[2];
    const float* pos     = (const float*)d_in[3];
    const float* enc_W0  = (const float*)d_in[4];
    const float* enc_b0  = (const float*)d_in[5];
    const float* enc_W1  = (const float*)d_in[6];
    const float* enc_b1  = (const float*)d_in[7];
    const float* enc_W2  = (const float*)d_in[8];
    const float* enc_b2  = (const float*)d_in[9];
    const float* qenc_W0 = (const float*)d_in[10];
    const float* qenc_b0 = (const float*)d_in[11];
    const float* qenc_W1 = (const float*)d_in[12];
    const float* qenc_b1 = (const float*)d_in[13];
    const float* qenc_W2 = (const float*)d_in[14];
    const float* qenc_b2 = (const float*)d_in[15];
    const float* dec_W0  = (const float*)d_in[16];
    const float* dec_b0  = (const float*)d_in[17];
    const float* dec_W1  = (const float*)d_in[18];
    const float* dec_b1  = (const float*)d_in[19];
    const float* dec_W2  = (const float*)d_in[20];
    const float* dec_b2  = (const float*)d_in[21];
    const float* msg_W   = (const float*)d_in[22];
    const float* msg_b   = (const float*)d_in[23];
    const float* ln1_g   = (const float*)d_in[24];
    const float* ln1_b   = (const float*)d_in[25];
    const float* node_W  = (const float*)d_in[26];
    const float* node_b  = (const float*)d_in[27];
    const float* ln2_g   = (const float*)d_in[28];
    const float* ln2_b   = (const float*)d_in[29];
    const int*   senders   = (const int*)d_in[30];
    const int*   receivers = (const int*)d_in[31];

    // workspace bump allocation (sc buffer reused for st, hA/hB for all MLPs)
    char*  base = (char*)d_ws;
    size_t off  = 0;
    auto alloc = [&](size_t bytes) { size_t o = off; off = (off + bytes + 255) & ~(size_t)255; return o; };
    _Float16* sc    = (_Float16*)(base + alloc(B * Nc * Nn * 2));   // scores (ctx, then tgt)
    _Float16* emb   = (_Float16*)(base + alloc(B * Nc * H_DIM * 2));
    float*    lat0  = (float*)(base + alloc(B * Nn * H_DIM * 4));
    float*    lat1  = (float*)(base + alloc(B * Nn * H_DIM * 4));
    float*    inbox = (float*)(base + alloc(B * Nn * H_DIM * 4));
    float*    zq    = (float*)(base + alloc(B * Nt * 256 * 4));     // [.,0:128]=z, [.,128:256]=q
    float*    hA    = (float*)(base + alloc(MR * H_DIM * 4));
    float*    hB    = (float*)(base + alloc(MR * H_DIM * 4));

    // ---- context scores ----
    scores_kernel<<<dim3(B * Nc / 8), dim3(256), 0, stream>>>(xc, pos, sc);

    // ---- context encoder MLP ----
    mlp0_kernel<<<dim3(MR * H_DIM / 256), dim3(256), 0, stream>>>(xc, 2, yc, 3, enc_W0, enc_b0, hA, (int)MR);
    {
        Epi e1 = epi_plain(hB, H_DIM); e1.bias = enc_b1; e1.relu = 1;
        launch_gemm(stream, AMatF32{hA, H_DIM}, BF32Row{enc_W1}, (int)MR, H_DIM, e1);
        Epi e2{enc_b2, nullptr, nullptr, nullptr, emb, H_DIM, 0, nullptr, nullptr};
        launch_gemm(stream, AMatF32{hB, H_DIM}, BF32Row{enc_W2}, (int)MR, H_DIM, e2);
    }

    // ---- latents = sc^T @ emb  (per batch) ----
    for (int b = 0; b < (int)B; ++b) {
        launch_gemm(stream,
                    AScT{sc + (size_t)b * Nc * Nn, (int)Nn},
                    BF16Row{emb + (size_t)b * Nc * H_DIM},
                    (int)Nn, (int)Nc,
                    epi_plain(lat0 + (size_t)b * Nn * H_DIM, H_DIM));
    }

    // ---- 4 message-passing steps (shared weights) ----
    float* cur = lat0;
    float* nxt = lat1;
    for (int step = 0; step < 4; ++step) {
        zero_f32_kernel<<<dim3(B * Nn * H_DIM / 256), dim3(256), 0, stream>>>(inbox, (int)(B * Nn * H_DIM));
        for (int b = 0; b < (int)B; ++b) {
            // m = LN1(concat(nodes[recv], nodes[send]) @ msg_W + msg_b); scatter-add into inbox
            Epi em{msg_b, ln1_g, ln1_b, nullptr, nullptr, H_DIM, 0,
                   receivers, inbox + (size_t)b * Nn * H_DIM};
            launch_gemm(stream,
                        AEdgeConcat{cur + (size_t)b * Nn * H_DIM, receivers, senders},
                        BF32Row{msg_W}, (int)E, 2 * H_DIM, em);
        }
        for (int b = 0; b < (int)B; ++b) {
            // nodes' = LN2(concat(nodes, inbox) @ node_W + node_b)
            Epi en{node_b, ln2_g, ln2_b, nxt + (size_t)b * Nn * H_DIM, nullptr, H_DIM, 0,
                   nullptr, nullptr};
            launch_gemm(stream,
                        AConcat2F32{cur + (size_t)b * Nn * H_DIM,
                                    inbox + (size_t)b * Nn * H_DIM, H_DIM},
                        BF32Row{node_W}, (int)Nn, 2 * H_DIM, en);
        }
        float* t = cur; cur = nxt; nxt = t;
    }
    // after 4 steps, final latents are in `cur` (== lat0)

    // ---- target scores (reuse sc buffer) + query MLP ----
    scores_kernel<<<dim3(B * Nt / 8), dim3(256), 0, stream>>>(xt, pos, sc);
    mlp0_kernel<<<dim3(MR * H_DIM / 256), dim3(256), 0, stream>>>(xt, 2, nullptr, 0, qenc_W0, qenc_b0, hA, (int)MR);
    {
        Epi e1 = epi_plain(hB, H_DIM); e1.bias = qenc_b1; e1.relu = 1;
        launch_gemm(stream, AMatF32{hA, H_DIM}, BF32Row{qenc_W1}, (int)MR, H_DIM, e1);
        Epi e2 = epi_plain(zq + H_DIM, 256); e2.bias = qenc_b2;      // q -> zq[:,128:256]
        launch_gemm(stream, AMatF32{hB, H_DIM}, BF32Row{qenc_W2}, (int)MR, H_DIM, e2);
    }

    // ---- z = st @ latents  (per batch) -> zq[:,0:128] ----
    for (int b = 0; b < (int)B; ++b) {
        launch_gemm(stream,
                    AF16Row{sc + (size_t)b * Nt * Nn, (int)Nn},
                    BF32Row{cur + (size_t)b * Nn * H_DIM},
                    (int)Nt, (int)Nn,
                    epi_plain(zq + (size_t)b * Nt * 256, 256));
    }

    // ---- decoder ----
    {
        Epi e0 = epi_plain(hA, H_DIM); e0.bias = dec_b0; e0.relu = 1;
        launch_gemm(stream, AMatF32{zq, 256}, BF32Row{dec_W0}, (int)MR, 256, e0);
        Epi e1 = epi_plain(hB, H_DIM); e1.bias = dec_b1; e1.relu = 1;
        launch_gemm(stream, AMatF32{hA, H_DIM}, BF32Row{dec_W1}, (int)MR, H_DIM, e1);
    }
    dec_final_kernel<<<dim3(MR / 256), dim3(256), 0, stream>>>(hB, dec_W2, dec_b2, (float*)d_out, (int)MR);
}
